// GVPGNNModel_69887707841294
// MI455X (gfx1250) — compile-verified, matrix operand order checked by reading, not run on hardware
//
#include <hip/hip_runtime.h>
#include <cstdint>
#include <cstddef>

// ---------------------------------------------------------------------------
// GVP-GNN forward for MI455X (gfx1250, wave32, WMMA).
//
// Input layout (jax pytree: dicts flatten in sorted-key order):
//   0: x [16384,64]   1: pos [16384,3]
//   params leaves (sorted keys: W_e, W_out, W_v, emb_b, emb_w, layers):
//   2..7   W_e  : wh[1,1], ws_b[32], ws_w[9,32], wsv_b[1], wsv_w[32,1], wv[1,1]
//   8..10  W_out: wh[16,16], ws_b[128], ws_w[144,128]
//   11..12 W_v  : ws_b[128], ws_w[128,128]
//   13 emb_b[128]  14 emb_w[64,128]
//   15+30L .. : layer L, gvp order f0,f1,m0,m1,m2; per-gvp leaf order
//               wh, ws_b, ws_w, wsv_b, wsv_w, wv
//   165: edge_index [2,131072] int32   166: batch [16384] int32
// Output: node_emb [16384,128] fp32 ++ graph_emb [64,128] fp32
// ---------------------------------------------------------------------------

typedef __bf16 v16bf __attribute__((ext_vector_type(16)));
typedef __bf16 v2bf  __attribute__((ext_vector_type(2)));
typedef float  v8f   __attribute__((ext_vector_type(8)));

#define NN 16384
#define NE 131072
#define NGR 64

// ---- input indices ---------------------------------------------------------
enum {
  IN_X = 0, IN_POS = 1,
  IW_E_WH = 2, IW_E_WSB, IW_E_WSW, IW_E_WSVB, IW_E_WSVW, IW_E_WV,
  IW_O_WH = 8, IW_O_WSB, IW_O_WSW,
  IW_V_WSB = 11, IW_V_WSW,
  IW_EMB_B = 13, IW_EMB_W = 14,
  IW_L0 = 15,          // +30 per layer; f0:+0 f1:+6 m0:+12 m1:+18 m2:+24
  IN_EI = 165, IN_BATCH = 166
};

// ---- workspace byte offsets ------------------------------------------------
constexpr size_t OFF_S   = 0;
constexpr size_t OFF_V   = OFF_S   + (size_t)NN * 128 * 4;
constexpr size_t OFF_DS  = OFF_V   + (size_t)NN * 48 * 4;
constexpr size_t OFF_DV  = OFF_DS  + (size_t)NN * 128 * 4;
constexpr size_t OFF_CNT = OFF_DV  + (size_t)NN * 48 * 4;
constexpr size_t OFF_SBF = OFF_CNT + (size_t)NN * 4;
constexpr size_t OFF_ESB = OFF_SBF + (size_t)NN * 128 * 2;
constexpr size_t OFF_EV  = OFF_ESB + (size_t)NE * 32 * 2;
constexpr size_t OFF_WT  = OFF_EV  + (size_t)NE * 3 * 4;

// packed-weight element offsets (bf16, [N][Kpad] N-major, K zero-padded to 32)
constexpr size_t WT_EMB = 0;                         // [128][64]
constexpr size_t WT_WV  = WT_EMB + 128 * 64;         // [128][128]
constexpr size_t WT_OUT = WT_WV  + 128 * 128;        // [128][160] (K=144)
constexpr size_t WT_L0  = WT_OUT + 128 * 160;
constexpr size_t WTL_M0_WS = 0;                      // [128][352] (K=321)
constexpr size_t WTL_M0_SV = WTL_M0_WS + 128 * 352;  // [16][128]
constexpr size_t WTL_M1_WS = WTL_M0_SV + 16 * 128;   // [128][160] (K=144)
constexpr size_t WTL_M1_SV = WTL_M1_WS + 128 * 160;
constexpr size_t WTL_M2_WS = WTL_M1_SV + 16 * 128;
constexpr size_t WTL_M2_SV = WTL_M2_WS + 128 * 160;
constexpr size_t WTL_F0_WS = WTL_M2_SV + 16 * 128;   // [512][160]
constexpr size_t WTL_F0_SV = WTL_F0_WS + 512 * 160;  // [32][512]
constexpr size_t WTL_F1_WS = WTL_F0_SV + 32 * 512;   // [128][544] (K=544 exact)
constexpr size_t WTL_F1_SV = WTL_F1_WS + 128 * 544;  // [16][128]
constexpr size_t WTL_SIZE  = WTL_F1_SV + 16 * 128;   // 262144 elements

// ---------------------------------------------------------------------------
// device helpers
// ---------------------------------------------------------------------------
__device__ inline float sigmoidf(float x) { return 1.f / (1.f + __expf(-x)); }

__device__ inline float grp_sum16(float v) {
#pragma unroll
  for (int m = 1; m < 16; m <<= 1) v += __shfl_xor(v, m, 32);
  return v;
}

// Block-level GEMM: out[16][SO] = A[16][Kpad] (bf16-convertible) x Wt + bias.
// Wt is bf16 N-major [SO][Kpad] (i.e. B transposed), Kpad % 32 == 0.
// 8 waves; wave w owns output column tiles n0 = 16w, 16w+128, ...
// Fragment layouts per CDNA5 ISA 7.12.2 (16-bit A 16x32 / 32-bit C 16x16).
template <typename T>
__device__ inline void block_gemm(const T* A, int Kpad, const __bf16* Wt,
                                  const float* bias, float* out, int SO) {
  const int lane = threadIdx.x & 31;
  const int wave = threadIdx.x >> 5;
  const int rc = lane & 15;          // A row / B col
  const int kh = (lane >> 4) * 8;    // K half select
  for (int n0 = wave * 16; n0 < SO; n0 += 128) {
    // warm this lane's weight panel into the near caches ahead of the K loop
    __builtin_prefetch(Wt + (size_t)(n0 + rc) * Kpad, 0, 1);
    v8f acc = {};
    for (int k0 = 0; k0 < Kpad; k0 += 32) {
      const T* pa = A + (size_t)rc * Kpad + k0 + kh;
      const __bf16* pb = Wt + (size_t)(n0 + rc) * Kpad + k0 + kh;
      v16bf a, b;
#pragma unroll
      for (int i = 0; i < 8; ++i) {
        a[i] = (__bf16)pa[i];
        a[i + 8] = (__bf16)pa[i + 16];
        b[i] = pb[i];
        b[i + 8] = pb[i + 16];
      }
      acc = __builtin_amdgcn_wmma_f32_16x16x32_bf16(false, a, false, b,
                                                    (short)0, acc, false, false);
    }
    const int col = n0 + rc;
    const float bs = bias ? bias[col] : 0.f;
#pragma unroll
    for (int i = 0; i < 8; ++i)
      out[(size_t)(i + (lane >> 4) * 8) * SO + col] = acc[i] + bs;
  }
}

// scalar LayerNorm of 16 rows x 128 cols in LDS; 16 threads per row.
__device__ inline void ln_s_lds(float* S) {
  const int g = threadIdx.x >> 4;
  const int li = threadIdx.x & 15;
  float s1 = 0.f, s2 = 0.f;
#pragma unroll
  for (int j = 0; j < 8; ++j) {
    float t = S[g * 128 + li * 8 + j];
    s1 += t; s2 += t * t;
  }
  s1 = grp_sum16(s1); s2 = grp_sum16(s2);
  float mu = s1 * (1.f / 128.f);
  float inv = rsqrtf(fmaxf(s2 * (1.f / 128.f) - mu * mu, 0.f) + 1e-5f);
#pragma unroll
  for (int j = 0; j < 8; ++j) {
    int k = g * 128 + li * 8 + j;
    S[k] = (S[k] - mu) * inv;
  }
}

// gvp.LayerNorm tuple: scalar LN (128) + vector RMS over 16 channels (x3).
__device__ inline void ln_tuple_lds(float* S, float* V) {
  ln_s_lds(S);
  const int g = threadIdx.x >> 4;
  const int li = threadIdx.x & 15;
  float ss = 0.f;
#pragma unroll
  for (int x = 0; x < 3; ++x) {
    float t = V[g * 48 + li * 3 + x];
    ss += t * t;
  }
  ss = fmaxf(ss, 1e-8f);                 // per-channel clip (norm_no_nan eps)
  float tot = grp_sum16(ss);
  float vninv = rsqrtf(tot * (1.f / 16.f));
#pragma unroll
  for (int x = 0; x < 3; ++x) V[g * 48 + li * 3 + x] *= vninv;
}

// ---------------------------------------------------------------------------
// small utility kernels
// ---------------------------------------------------------------------------
__global__ void k_zero(float* p, int n) {
  int i = blockIdx.x * 256 + threadIdx.x;
  if (i < n) p[i] = 0.f;
}

__global__ void k_pack(const float* src, __bf16* dst, int K, int N, int Kpad) {
  int i = blockIdx.x * 256 + threadIdx.x;
  if (i >= N * Kpad) return;
  int n = i / Kpad, k = i % Kpad;
  dst[i] = (k < K) ? (__bf16)src[(size_t)k * N + n] : (__bf16)0.f;
}

__global__ void k_count(const int* ei, float* cnt) {
  int e = blockIdx.x * 256 + threadIdx.x;
  if (e < NE) atomicAdd(&cnt[ei[NE + e]], 1.f);
}

// ---------------------------------------------------------------------------
// edge embedding: radial bessel * poly cutoff -> LN tuple -> W_e GVP (9->32,1->1)
// ---------------------------------------------------------------------------
struct EdgeArgs {
  const float* pos; const int* ei;
  const float *wh, *wsb, *wsw, *wsvb, *wsvw, *wv;
  __bf16* es_bf; float* ev;
};

__global__ __launch_bounds__(256) void k_edge(EdgeArgs a) {
  int e = blockIdx.x * 256 + threadIdx.x;
  if (e >= NE) return;
  int s = a.ei[e], t = a.ei[NE + e];
  float dx = a.pos[s * 3 + 0] - a.pos[t * 3 + 0];
  float dy = a.pos[s * 3 + 1] - a.pos[t * 3 + 1];
  float dz = a.pos[s * 3 + 2] - a.pos[t * 3 + 2];
  float len = sqrtf(dx * dx + dy * dy + dz * dz);
  float il = (len > 0.f) ? 1.f / fmaxf(len, 1e-12f) : 0.f;
  float ux = dx * il, uy = dy * il, uz = dz * il;
  // radial: sqrt(2/R)*sin(n*pi*len/R)/len * env(u), p=5 cutoff
  float u = len * 0.1f;
  float u5 = u * u * u * u * u;
  float env = 1.f - 21.f * u5 + 35.f * u5 * u - 15.f * u5 * u * u;
  env *= (len < 10.f) ? 1.f : 0.f;
  float pref = 0.4472135955f / fmaxf(len, 1e-12f);  // sqrt(0.2)/len
  float rb[8];
#pragma unroll
  for (int n = 0; n < 8; ++n)
    rb[n] = pref * __sinf((float)(n + 1) * 0.314159265358979f * len) * env;
  // scalar LN over 8
  float mu = 0.f;
#pragma unroll
  for (int n = 0; n < 8; ++n) mu += rb[n];
  mu *= 0.125f;
  float var = 0.f;
#pragma unroll
  for (int n = 0; n < 8; ++n) { float d = rb[n] - mu; var += d * d; }
  float inv = rsqrtf(var * 0.125f + 1e-5f);
#pragma unroll
  for (int n = 0; n < 8; ++n) rb[n] = (rb[n] - mu) * inv;
  // vector RMS over 1 channel
  float vn = sqrtf(fmaxf(ux * ux + uy * uy + uz * uz, 1e-8f));
  float wx = ux / vn, wy = uy / vn, wz = uz / vn;
  // GVP (si=8, vi=1, h=1, so=32, vo=1), vector_gate
  float wh00 = a.wh[0], wv00 = a.wv[0];
  float hx = wx * wh00, hy = wy * wh00, hz = wz * wh00;
  float vnh = sqrtf(fmaxf(hx * hx + hy * hy + hz * hz, 1e-8f));
  float gdot = a.wsvb[0];
  float esl[32];
  for (int o = 0; o < 32; ++o) {
    float acc = a.wsb[o] + vnh * a.wsw[8 * 32 + o];
#pragma unroll
    for (int k = 0; k < 8; ++k) acc += rb[k] * a.wsw[k * 32 + o];
    esl[o] = acc;
    gdot += acc * a.wsvw[o];
  }
#pragma unroll
  for (int o = 0; o < 32; o += 2) {      // 32-bit packed bf16 stores
    v2bf p;
    p[0] = (__bf16)esl[o];
    p[1] = (__bf16)esl[o + 1];
    *(v2bf*)&a.es_bf[(size_t)e * 32 + o] = p;
  }
  float g = sigmoidf(gdot);
  a.ev[(size_t)e * 3 + 0] = hx * wv00 * g;
  a.ev[(size_t)e * 3 + 1] = hy * wv00 * g;
  a.ev[(size_t)e * 3 + 2] = hz * wv00 * g;
}

// ---------------------------------------------------------------------------
// node embedding: x@emb_w + b -> LN -> W_v (128->128); v := 0
// ---------------------------------------------------------------------------
struct EmbArgs {
  const float* x; const __bf16* wt_emb; const float* emb_b;
  const __bf16* wt_wv; const float* wv_b;
  float* S; __bf16* SBF; float* V;
};

__global__ __launch_bounds__(256) void k_node_embed(EmbArgs a) {
  __shared__ float eS[16 * 128];
  __shared__ float eO[16 * 128];
  const int n0 = blockIdx.x * 16;
  const int tid = threadIdx.x;
  block_gemm<float>(a.x + (size_t)n0 * 64, 64, a.wt_emb, a.emb_b, eS, 128);
  __syncthreads();
  ln_s_lds(eS);
  __syncthreads();
  block_gemm<float>(eS, 128, a.wt_wv, a.wv_b, eO, 128);
  __syncthreads();
  for (int i = tid; i < 16 * 64; i += 256) {
    int e = i >> 6, c2 = (i & 63) * 2;
    int n = n0 + e;
    float v0 = eO[e * 128 + c2], v1 = eO[e * 128 + c2 + 1];
    a.S[(size_t)n * 128 + c2] = v0;
    a.S[(size_t)n * 128 + c2 + 1] = v1;
    v2bf p; p[0] = (__bf16)v0; p[1] = (__bf16)v1;
    *(v2bf*)&a.SBF[(size_t)n * 128 + c2] = p;
  }
  for (int i = tid; i < 16 * 48; i += 256)
    a.V[(size_t)(n0 + i / 48) * 48 + (i % 48)] = 0.f;
}

// ---------------------------------------------------------------------------
// fused message kernel: per 16-edge tile, runs m0->m1->m2 GVP chain in LDS,
// scatters results via fp32 atomics into ds/dv.
// LDS vector buffer layout (stride 100 floats, 16B aligned rows):
//   channels 0..15 = v_j, 16..31 = v_i, 32 = ev  (ref wh order is v_j|ev|v_i,
//   handled by index remap in the m0 stage).
// ---------------------------------------------------------------------------
struct MsgArgs {
  const __bf16* s_bf; const float* v; const __bf16* es_bf; const float* ev;
  const int* ei; float* ds; float* dv;
  const float *m0_wh, *m0_wsb, *m0_wsvb, *m0_wv;
  const __bf16 *m0_ws, *m0_sv;
  const float *m1_wh, *m1_wsb, *m1_wsvb, *m1_wv;
  const __bf16 *m1_ws, *m1_sv;
  const float *m2_wh, *m2_wsb, *m2_wsvb, *m2_wv;
  const __bf16 *m2_ws, *m2_sv;
};

__device__ void msg_stage(int vi, int h, int Sbase, int Kpad, int KpadNext,
                          bool relu_out, bool remap, bool last,
                          const float* wh, const __bf16* wt_ws, const float* wsb,
                          const __bf16* wt_sv, const float* wsvb, const float* wv,
                          __bf16* lsA, float* lsS, float* lsMV, float* lsVH,
                          float* lsG, const int* lsTgt, float* ds, float* dv) {
  const int tid = threadIdx.x;
  // vh = mv @ wh  [16][h][3]  (VALU, K<=33)
  for (int i = tid; i < 16 * h * 3; i += 256) {
    int e = i / (h * 3), r = i % (h * 3), hh = r / 3, x = r % 3;
    float acc = 0.f;
    for (int c = 0; c < vi; ++c) {
      int cr = remap ? ((c < 16) ? c : (c < 32 ? c + 1 : 16)) : c;
      acc += lsMV[e * 100 + c * 3 + x] * wh[cr * h + hh];
    }
    lsVH[e * 100 + hh * 3 + x] = acc;
  }
  __syncthreads();
  // vn -> scalar tail of A, zero pad to Kpad
  const int tail = Kpad - Sbase;
  for (int i = tid; i < 16 * tail; i += 256) {
    int e = i / tail, c = i % tail;
    float val = 0.f;
    if (c < h) {
      float ss = 0.f;
#pragma unroll
      for (int x = 0; x < 3; ++x) {
        float t = lsVH[e * 100 + c * 3 + x];
        ss += t * t;
      }
      val = sqrtf(fmaxf(ss, 1e-8f));
    }
    lsA[e * Kpad + Sbase + c] = (__bf16)val;
  }
  __syncthreads();
  block_gemm<__bf16>(lsA, Kpad, wt_ws, wsb, lsS, 128);   // pre-activation s_out
  __syncthreads();
  block_gemm<float>(lsS, 128, wt_sv, wsvb, lsG, 16);     // gate logits (wave 0)
  __syncthreads();
  // v_out = (vh @ wv) * sigmoid(gate) -> lsMV channels 0..15
  for (int i = tid; i < 16 * 16 * 3; i += 256) {
    int e = i / 48, r = i % 48, c = r / 3, x = r % 3;
    float acc = 0.f;
    for (int hh = 0; hh < h; ++hh)
      acc += lsVH[e * 100 + hh * 3 + x] * wv[hh * 16 + c];
    lsMV[e * 100 + c * 3 + x] = acc * sigmoidf(lsG[e * 16 + c]);
  }
  if (!last) {
    for (int i = tid; i < 16 * 128; i += 256) {
      int e = i >> 7, c = i & 127;
      float sv = lsS[e * 128 + c];
      if (relu_out) sv = fmaxf(sv, 0.f);
      lsA[e * KpadNext + c] = (__bf16)sv;
    }
  } else {
    for (int i = tid; i < 16 * 128; i += 256) {
      int e = i >> 7, c = i & 127;
      atomicAdd(&ds[(size_t)lsTgt[e] * 128 + c], lsS[e * 128 + c]);
    }
  }
  __syncthreads();
  if (last) {
    for (int i = tid; i < 16 * 48; i += 256) {
      int e = i / 48, r = i % 48;
      atomicAdd(&dv[(size_t)lsTgt[e] * 48 + r], lsMV[e * 100 + r]);
    }
    __syncthreads();
  }
}

__global__ __launch_bounds__(256) void k_msg(MsgArgs a) {
  __shared__ __align__(16) __bf16 lsA[16 * 352];
  __shared__ float lsS[16 * 128];
  __shared__ __align__(16) float lsMV[16 * 100];
  __shared__ float lsVH[16 * 100];
  __shared__ float lsG[16 * 16];
  __shared__ int lsSrc[16], lsTgt[16];
  const int tid = threadIdx.x;
  const int e0 = blockIdx.x * 16;
  if (tid < 16) {
    lsSrc[tid] = a.ei[e0 + tid];
    lsTgt[tid] = a.ei[NE + e0 + tid];
  }
  __syncthreads();
  // gather scalar concat [s_j | es | s_i] into A with 16B vector moves
  for (int i = tid; i < 16 * 16; i += 256) {        // s_j: 16 chunks x 8 bf16
    int e = i >> 4, cc = i & 15;
    *(uint4*)&lsA[e * 352 + cc * 8] =
        *(const uint4*)&a.s_bf[(size_t)lsSrc[e] * 128 + cc * 8];
  }
  for (int i = tid; i < 16 * 16; i += 256) {        // s_i at col 160
    int e = i >> 4, cc = i & 15;
    *(uint4*)&lsA[e * 352 + 160 + cc * 8] =
        *(const uint4*)&a.s_bf[(size_t)lsTgt[e] * 128 + cc * 8];
  }
  for (int i = tid; i < 16 * 4; i += 256) {         // es at col 128
    int e = i >> 2, cc = i & 3;
    *(uint4*)&lsA[e * 352 + 128 + cc * 8] =
        *(const uint4*)&a.es_bf[(size_t)(e0 + e) * 32 + cc * 8];
  }
  // gather vectors: channels 0..15 = v_j, 16..31 = v_i, 32 = ev
  for (int i = tid; i < 16 * 12; i += 256) {        // v_j: 12 chunks x 4 f32
    int e = i / 12, cc = i % 12;
    *(uint4*)&lsMV[e * 100 + cc * 4] =
        *(const uint4*)&a.v[(size_t)lsSrc[e] * 48 + cc * 4];
  }
  for (int i = tid; i < 16 * 12; i += 256) {        // v_i at float 48
    int e = i / 12, cc = i % 12;
    *(uint4*)&lsMV[e * 100 + 48 + cc * 4] =
        *(const uint4*)&a.v[(size_t)lsTgt[e] * 48 + cc * 4];
  }
  for (int i = tid; i < 16 * 3; i += 256) {         // ev at float 96
    int e = i / 3, x = i % 3;
    lsMV[e * 100 + 96 + x] = a.ev[(size_t)(e0 + e) * 3 + x];
  }
  __syncthreads();
  msg_stage(33, 33, 288, 352, 160, true, true, false, a.m0_wh, a.m0_ws,
            a.m0_wsb, a.m0_sv, a.m0_wsvb, a.m0_wv, lsA, lsS, lsMV, lsVH, lsG,
            lsTgt, a.ds, a.dv);
  msg_stage(16, 16, 128, 160, 160, true, false, false, a.m1_wh, a.m1_ws,
            a.m1_wsb, a.m1_sv, a.m1_wsvb, a.m1_wv, lsA, lsS, lsMV, lsVH, lsG,
            lsTgt, a.ds, a.dv);
  msg_stage(16, 16, 128, 160, 160, false, false, true, a.m2_wh, a.m2_ws,
            a.m2_wsb, a.m2_sv, a.m2_wsvb, a.m2_wv, lsA, lsS, lsMV, lsVH, lsG,
            lsTgt, a.ds, a.dv);
}

// ---------------------------------------------------------------------------
// fused node update: mean-aggregate + residual + LN + FFN(f0,f1) + residual + LN
// ---------------------------------------------------------------------------
struct UpdArgs {
  float* S; float* V; const float* DS; const float* DV; const float* CNT;
  __bf16* SBF;
  const float *f0_wh, *f0_wsb, *f0_wsvb, *f0_wv;
  const __bf16 *f0_ws, *f0_sv;
  const float *f1_wh, *f1_wsb, *f1_wsvb, *f1_wv;
  const __bf16 *f1_ws, *f1_sv;
};

__global__ __launch_bounds__(256) void k_node_update(UpdArgs a) {
  __shared__ float nS[16 * 128];
  __shared__ float nV[16 * 48];
  __shared__ float nV2[16 * 96];
  __shared__ float nVH[16 * 96];
  __shared__ float nS2[16 * 512];
  __shared__ float nFS[16 * 128];
  __shared__ float nG[16 * 32];
  __shared__ __align__(16) __bf16 nA[16 * 544];
  const int n0 = blockIdx.x * 16;
  const int tid = threadIdx.x;
  // 1. residual + mean aggregation
  for (int i = tid; i < 16 * 128; i += 256) {
    int n = n0 + (i >> 7), c = i & 127;
    float cn = fmaxf(a.CNT[n], 1.f);
    nS[i] = a.S[(size_t)n * 128 + c] + a.DS[(size_t)n * 128 + c] / cn;
  }
  for (int i = tid; i < 16 * 48; i += 256) {
    int n = n0 + i / 48, r = i % 48;
    float cn = fmaxf(a.CNT[n], 1.f);
    nV[i] = a.V[(size_t)n * 48 + r] + a.DV[(size_t)n * 48 + r] / cn;
  }
  __syncthreads();
  ln_tuple_lds(nS, nV);
  __syncthreads();
  // 3. f0: vh0 = v @ wh_f0 [16,32]
  for (int i = tid; i < 16 * 32 * 3; i += 256) {
    int e = i / 96, r = i % 96, hh = r / 3, x = r % 3;
    float acc = 0.f;
    for (int c = 0; c < 16; ++c) acc += nV[e * 48 + c * 3 + x] * a.f0_wh[c * 32 + hh];
    nVH[e * 96 + hh * 3 + x] = acc;
  }
  __syncthreads();
  for (int i = tid; i < 16 * 160; i += 256) {
    int e = i / 160, c = i % 160;
    float val;
    if (c < 128) val = nS[e * 128 + c];
    else {
      int hh = c - 128;
      float ss = 0.f;
#pragma unroll
      for (int x = 0; x < 3; ++x) { float t = nVH[e * 96 + hh * 3 + x]; ss += t * t; }
      val = sqrtf(fmaxf(ss, 1e-8f));
    }
    nA[e * 160 + c] = (__bf16)val;
  }
  __syncthreads();
  block_gemm<__bf16>(nA, 160, a.f0_ws, a.f0_wsb, nS2, 512);
  __syncthreads();
  block_gemm<float>(nS2, 512, a.f0_sv, a.f0_wsvb, nG, 32);
  __syncthreads();
  // v_f0 = (vh0 @ wv_f0) * sigmoid(gate)
  for (int i = tid; i < 16 * 32 * 3; i += 256) {
    int e = i / 96, r = i % 96, c = r / 3, x = r % 3;
    float acc = 0.f;
    for (int hh = 0; hh < 32; ++hh)
      acc += nVH[e * 96 + hh * 3 + x] * a.f0_wv[hh * 32 + c];
    nV2[e * 96 + r] = acc * sigmoidf(nG[e * 32 + c]);
  }
  __syncthreads();
  // f1: vh1 = v_f0 @ wh_f1 [32,32]
  for (int i = tid; i < 16 * 32 * 3; i += 256) {
    int e = i / 96, r = i % 96, hh = r / 3, x = r % 3;
    float acc = 0.f;
    for (int c = 0; c < 32; ++c) acc += nV2[e * 96 + c * 3 + x] * a.f1_wh[c * 32 + hh];
    nVH[e * 96 + hh * 3 + x] = acc;
  }
  __syncthreads();
  for (int i = tid; i < 16 * 544; i += 256) {
    int e = i / 544, c = i % 544;
    float val;
    if (c < 512) val = fmaxf(nS2[e * 512 + c], 0.f);   // f0 scalar_act = relu
    else {
      int hh = c - 512;
      float ss = 0.f;
#pragma unroll
      for (int x = 0; x < 3; ++x) { float t = nVH[e * 96 + hh * 3 + x]; ss += t * t; }
      val = sqrtf(fmaxf(ss, 1e-8f));
    }
    nA[e * 544 + c] = (__bf16)val;
  }
  __syncthreads();
  block_gemm<__bf16>(nA, 544, a.f1_ws, a.f1_wsb, nFS, 128);
  __syncthreads();
  block_gemm<float>(nFS, 128, a.f1_sv, a.f1_wsvb, nG, 16);
  __syncthreads();
  // residual: v += (vh1 @ wv_f1)*gate ; s += fs
  for (int i = tid; i < 16 * 48; i += 256) {
    int e = i / 48, r = i % 48, c = r / 3, x = r % 3;
    float acc = 0.f;
    for (int hh = 0; hh < 32; ++hh)
      acc += nVH[e * 96 + hh * 3 + x] * a.f1_wv[hh * 16 + c];
    nV[i] += acc * sigmoidf(nG[e * 16 + c]);
  }
  for (int i = tid; i < 16 * 128; i += 256) nS[i] += nFS[i];
  __syncthreads();
  ln_tuple_lds(nS, nV);
  __syncthreads();
  for (int i = tid; i < 16 * 64; i += 256) {
    int e = i >> 6, c2 = (i & 63) * 2;
    int n = n0 + e;
    float v0 = nS[e * 128 + c2], v1 = nS[e * 128 + c2 + 1];
    a.S[(size_t)n * 128 + c2] = v0;
    a.S[(size_t)n * 128 + c2 + 1] = v1;
    v2bf p; p[0] = (__bf16)v0; p[1] = (__bf16)v1;
    *(v2bf*)&a.SBF[(size_t)n * 128 + c2] = p;
  }
  for (int i = tid; i < 16 * 48; i += 256)
    a.V[(size_t)(n0 + i / 48) * 48 + (i % 48)] = nV[i];
}

// ---------------------------------------------------------------------------
// output head: LN tuple -> W_out GVP (144->128, relu) -> node_emb + sum pool
// ---------------------------------------------------------------------------
struct OutArgs {
  const float* S; const float* V; const int* batch;
  const float *wh, *wsb; const __bf16* wt;
  float* out;
};

__global__ __launch_bounds__(256) void k_out(OutArgs a) {
  __shared__ float oS[16 * 128];
  __shared__ float oV[16 * 48];
  __shared__ float oVH[16 * 48];
  __shared__ __align__(16) __bf16 oA[16 * 160];
  __shared__ float oO[16 * 128];
  const int n0 = blockIdx.x * 16;
  const int tid = threadIdx.x;
  for (int i = tid; i < 16 * 128; i += 256)
    oS[i] = a.S[(size_t)(n0 + (i >> 7)) * 128 + (i & 127)];
  for (int i = tid; i < 16 * 48; i += 256)
    oV[i] = a.V[(size_t)(n0 + i / 48) * 48 + (i % 48)];
  __syncthreads();
  ln_tuple_lds(oS, oV);
  __syncthreads();
  for (int i = tid; i < 16 * 16 * 3; i += 256) {
    int e = i / 48, r = i % 48, hh = r / 3, x = r % 3;
    float acc = 0.f;
    for (int c = 0; c < 16; ++c) acc += oV[e * 48 + c * 3 + x] * a.wh[c * 16 + hh];
    oVH[e * 48 + hh * 3 + x] = acc;
  }
  __syncthreads();
  for (int i = tid; i < 16 * 160; i += 256) {
    int e = i / 160, c = i % 160;
    float val = 0.f;
    if (c < 128) val = oS[e * 128 + c];
    else if (c < 144) {
      int hh = c - 128;
      float ss = 0.f;
#pragma unroll
      for (int x = 0; x < 3; ++x) { float t = oVH[e * 48 + hh * 3 + x]; ss += t * t; }
      val = sqrtf(fmaxf(ss, 1e-8f));
    }
    oA[e * 160 + c] = (__bf16)val;
  }
  __syncthreads();
  block_gemm<__bf16>(oA, 160, a.wt, a.wsb, oO, 128);
  __syncthreads();
  for (int i = tid; i < 16 * 128; i += 256) {
    int e = i >> 7, c = i & 127, n = n0 + e;
    float r = fmaxf(oO[i], 0.f);
    a.out[(size_t)n * 128 + c] = r;
    atomicAdd(&a.out[(size_t)NN * 128 + (size_t)a.batch[n] * 128 + c], r);
  }
}

// ---------------------------------------------------------------------------
// host launcher
// ---------------------------------------------------------------------------
extern "C" void kernel_launch(void* const* d_in, const int* in_sizes, int n_in,
                              void* d_out, int out_size, void* d_ws, size_t ws_size,
                              hipStream_t stream) {
  (void)in_sizes; (void)n_in; (void)out_size; (void)ws_size;
  auto F = [&](int i) { return (const float*)d_in[i]; };
  const int* ei = (const int*)d_in[IN_EI];
  const int* batch = (const int*)d_in[IN_BATCH];
  char* ws = (char*)d_ws;
  float* S = (float*)(ws + OFF_S);
  float* V = (float*)(ws + OFF_V);
  float* DS = (float*)(ws + OFF_DS);
  float* DV = (float*)(ws + OFF_DV);
  float* CNT = (float*)(ws + OFF_CNT);
  __bf16* SBF = (__bf16*)(ws + OFF_SBF);
  __bf16* ESB = (__bf16*)(ws + OFF_ESB);
  float* EV = (float*)(ws + OFF_EV);
  __bf16* WT = (__bf16*)(ws + OFF_WT);
  float* OUT = (float*)d_out;

  auto pack = [&](int idx, __bf16* dst, int K, int N, int Kpad) {
    int n = N * Kpad;
    k_pack<<<(n + 255) / 256, 256, 0, stream>>>(F(idx), dst, K, N, Kpad);
  };

  // pack weights (bf16, transposed, K padded to 32)
  pack(IW_EMB_W, WT + WT_EMB, 64, 128, 64);
  pack(IW_V_WSW, WT + WT_WV, 128, 128, 128);
  pack(IW_O_WSW, WT + WT_OUT, 144, 128, 160);
  for (int L = 0; L < 5; ++L) {
    int b = IW_L0 + 30 * L;
    __bf16* wl = WT + WT_L0 + (size_t)L * WTL_SIZE;
    pack(b + 14, wl + WTL_M0_WS, 321, 128, 352);
    pack(b + 16, wl + WTL_M0_SV, 128, 16, 128);
    pack(b + 20, wl + WTL_M1_WS, 144, 128, 160);
    pack(b + 22, wl + WTL_M1_SV, 128, 16, 128);
    pack(b + 26, wl + WTL_M2_WS, 144, 128, 160);
    pack(b + 28, wl + WTL_M2_SV, 128, 16, 128);
    pack(b + 2,  wl + WTL_F0_WS, 160, 512, 160);
    pack(b + 4,  wl + WTL_F0_SV, 512, 32, 512);
    pack(b + 8,  wl + WTL_F1_WS, 544, 128, 544);
    pack(b + 10, wl + WTL_F1_SV, 128, 16, 128);
  }

  // init: counts + graph-pool region
  k_zero<<<(NN + 255) / 256, 256, 0, stream>>>(CNT, NN);
  k_zero<<<(NGR * 128 + 255) / 256, 256, 0, stream>>>(OUT + (size_t)NN * 128, NGR * 128);
  k_count<<<NE / 256, 256, 0, stream>>>(ei, CNT);

  // edge embedding
  EdgeArgs ea{F(IN_POS), ei, F(IW_E_WH), F(IW_E_WSB), F(IW_E_WSW),
              F(IW_E_WSVB), F(IW_E_WSVW), F(IW_E_WV), ESB, EV};
  k_edge<<<NE / 256, 256, 0, stream>>>(ea);

  // node embedding
  EmbArgs na{F(IN_X), WT + WT_EMB, F(IW_EMB_B), WT + WT_WV, F(IW_V_WSB), S, SBF, V};
  k_node_embed<<<NN / 16, 256, 0, stream>>>(na);

  // layers
  for (int L = 0; L < 5; ++L) {
    int b = IW_L0 + 30 * L;
    __bf16* wl = WT + WT_L0 + (size_t)L * WTL_SIZE;
    k_zero<<<(NN * 128 + 255) / 256, 256, 0, stream>>>(DS, NN * 128);
    k_zero<<<(NN * 48 + 255) / 256, 256, 0, stream>>>(DV, NN * 48);
    MsgArgs ma;
    ma.s_bf = SBF; ma.v = V; ma.es_bf = ESB; ma.ev = EV; ma.ei = ei;
    ma.ds = DS; ma.dv = DV;
    ma.m0_wh = F(b + 12); ma.m0_wsb = F(b + 13); ma.m0_wsvb = F(b + 15);
    ma.m0_wv = F(b + 17); ma.m0_ws = wl + WTL_M0_WS; ma.m0_sv = wl + WTL_M0_SV;
    ma.m1_wh = F(b + 18); ma.m1_wsb = F(b + 19); ma.m1_wsvb = F(b + 21);
    ma.m1_wv = F(b + 23); ma.m1_ws = wl + WTL_M1_WS; ma.m1_sv = wl + WTL_M1_SV;
    ma.m2_wh = F(b + 24); ma.m2_wsb = F(b + 25); ma.m2_wsvb = F(b + 27);
    ma.m2_wv = F(b + 29); ma.m2_ws = wl + WTL_M2_WS; ma.m2_sv = wl + WTL_M2_SV;
    k_msg<<<NE / 16, 256, 0, stream>>>(ma);
    UpdArgs ua;
    ua.S = S; ua.V = V; ua.DS = DS; ua.DV = DV; ua.CNT = CNT; ua.SBF = SBF;
    ua.f0_wh = F(b + 0); ua.f0_wsb = F(b + 1); ua.f0_wsvb = F(b + 3);
    ua.f0_wv = F(b + 5); ua.f0_ws = wl + WTL_F0_WS; ua.f0_sv = wl + WTL_F0_SV;
    ua.f1_wh = F(b + 6); ua.f1_wsb = F(b + 7); ua.f1_wsvb = F(b + 9);
    ua.f1_wv = F(b + 11); ua.f1_ws = wl + WTL_F1_WS; ua.f1_sv = wl + WTL_F1_SV;
    k_node_update<<<NN / 16, 256, 0, stream>>>(ua);
  }

  // output head + graph sum-pool
  OutArgs oa{S, V, batch, F(IW_O_WH), F(IW_O_WSB), WT + WT_OUT, OUT};
  k_out<<<NN / 16, 256, 0, stream>>>(oa);
}
